// NerfModel_43276090474699
// MI455X (gfx1250) — matrix operand order repaired and checked by reading, not compiled, not run on hardware
//
#include <hip/hip_runtime.h>
#include <hip/hip_fp16.h>

typedef __attribute__((ext_vector_type(16))) _Float16 v16h;
typedef __attribute__((ext_vector_type(8)))  _Float16 v8h;
typedef __attribute__((ext_vector_type(4)))  _Float16 v4h;
typedef __attribute__((ext_vector_type(8)))  float    v8f;

#define WAVES    8
#define LDSTRIDE 96

__device__ __forceinline__ v16h cat8(v8h lo, v8h hi) {
    return __builtin_shufflevector(lo, hi, 0,1,2,3,4,5,6,7,8,9,10,11,12,13,14,15);
}

__device__ __forceinline__ void lds_fence() {
    asm volatile("s_wait_dscnt 0x0" ::: "memory");
}

// ---------------------------------------------------------------------------
// Weight prep: convert f32 weights to f16 and lay out as WMMA B-fragments.
// Fragment (kc,nt) = 512 f16: lane l holds 16 consecutive K values
//   K = kc*32 + (l>>4)*16 + j   (j = 0..15),  N = nt*16 + (l&15)
// so the main kernel loads a B fragment as two coalesced b128 loads.
// Out-of-range (K>=Ksrc or N>=Nsrc) entries are zero (K/N padding).
// ---------------------------------------------------------------------------
__global__ void pack_weights(const float* __restrict__ src, int Ksrc, int Nsrc,
                             int KC, int NT, _Float16* __restrict__ dst) {
    const int e = blockIdx.x * blockDim.x + threadIdx.x;
    const int total = KC * NT * 512;
    if (e >= total) return;
    const int frag = e >> 9;
    const int r    = e & 511;
    const int lane = r >> 4;
    const int j    = r & 15;
    const int kc   = frag / NT;
    const int nt   = frag % NT;
    const int k    = kc * 32 + ((lane >> 4) << 4) + j;
    const int n    = nt * 16 + (lane & 15);
    const float v  = (k < Ksrc && n < Nsrc) ? src[k * Nsrc + n] : 0.0f;
    dst[e] = (_Float16)v;
}

// ---------------------------------------------------------------------------
// Wave-level tiled matmul: D[16, NT*16] += A_lds[16, KC*32] x Bpack
// A is in LDS, row-major f16 with row stride LDSTRIDE.
// ---------------------------------------------------------------------------
template <int KC, int NT>
__device__ __forceinline__ void wave_mm(const _Float16* Alds, const _Float16* Bpack,
                                        int lane, v8f acc[NT]) {
    const int m  = lane & 15;
    const int kh = (lane >> 4) << 4;  // 0 or 16 (A K-half select)
#pragma unroll
    for (int kc = 0; kc < KC; ++kc) {
        const _Float16* ar = Alds + m * LDSTRIDE + kc * 32 + kh;
        const v16h a = cat8(*(const v8h*)ar, *(const v8h*)(ar + 8));
#pragma unroll
        for (int nt = 0; nt < NT; ++nt) {
            const _Float16* br = Bpack + (((kc * NT + nt) << 9) + (lane << 4));
            const v16h b = cat8(*(const v8h*)br, *(const v8h*)(br + 8));
            acc[nt] = __builtin_amdgcn_wmma_f32_16x16x32_f16(
                false, a, false, b, (short)0, acc[nt], false, false);
        }
    }
}

// ---------------------------------------------------------------------------
// Fused NeRF kernel: one wave == 16 points.
// ---------------------------------------------------------------------------
__global__ __launch_bounds__(256)
void nerf_fused(const float* __restrict__ x,  const float* __restrict__ dvec,
                const float* __restrict__ xy, const float* __restrict__ yz,
                const float* __restrict__ xz,
                const float* __restrict__ b1, const float* __restrict__ b2,
                const float* __restrict__ c1, const float* __restrict__ c2,
                const float* __restrict__ c3,
                const _Float16* __restrict__ wp,
                float* __restrict__ outC, float* __restrict__ outS, int nB)
{
    __shared__ _Float16 sA[WAVES][16 * LDSTRIDE];
    __shared__ float    sMask[WAVES][16];

    const int  lane  = threadIdx.x & 31;
    const int  wv    = threadIdx.x >> 5;
    const long gbase = ((long)blockIdx.x * WAVES + wv) * 16L;
    if (gbase >= nB) return;           // wave-uniform: EXEC stays all-1 for WMMA

    _Float16* A   = sA[wv];
    float*    Msk = sMask[wv];

    // ---- per-point indices; both lane-halves duplicate point p = lane&15 ----
    const int p = lane & 15;
    long gp = gbase + p; if (gp >= nB) gp = nB - 1;
    const float x0 = x[gp * 3 + 0], x1 = x[gp * 3 + 1], x2 = x[gp * 3 + 2];
    const bool mask = (fabsf(x0) < 1000.0f) & (fabsf(x1) < 380.0f) & (fabsf(x2) < 150.0f);
    auto idx = [](float v, float sc) -> int {
        int i = (int)((v / (2.0f * sc) + 0.5f) * 512.0f);   // trunc toward zero
        return i < 0 ? 0 : (i > 511 ? 511 : i);
    };
    const int ix = idx(x0, 1000.0f), iy = idx(x1, 380.0f);
    const int jy = idx(x1, 380.0f),  jz = idx(x2, 150.0f);
    const int kx = idx(x1, 1000.0f), kz = idx(x2, 150.0f);   // faithful source bug
    const unsigned oXY = ((unsigned)ix * 512u + (unsigned)iy) * 96u;
    const unsigned oYZ = ((unsigned)jy * 512u + (unsigned)jz) * 96u;
    const unsigned oXZ = ((unsigned)kx * 512u + (unsigned)kz) * 96u;
    if (lane < 16) Msk[p] = mask ? 1.0f : 0.0f;

    // ---- tri-plane gather + fuse: each lane-half covers 48 features ----
    const int f0 = (lane >> 4) * 48;
#pragma unroll
    for (int it = 0; it < 12; ++it) {
        const int f = f0 + it * 4;
        const float4 a4 = *(const float4*)(xy + oXY + f);
        const float4 b4 = *(const float4*)(yz + oYZ + f);
        const float4 g4 = *(const float4*)(xz + oXZ + f);
        v4h t;
        t.x = (_Float16)(a4.x * b4.x * g4.x);
        t.y = (_Float16)(a4.y * b4.y * g4.y);
        t.z = (_Float16)(a4.z * b4.z * g4.z);
        t.w = (_Float16)(a4.w * b4.w * g4.w);
        *(v4h*)(A + p * LDSTRIDE + f) = t;
    }
    lds_fence();

    const int n   = lane & 15;          // D fragment: N = lane&15
    const int mhi = (lane >> 4) << 3;   // D fragment: M = r + 8*(lane>=16)
    const _Float16* W1P = wp;
    const _Float16* W2P = wp + 6144;
    const _Float16* V1P = wp + 7168;
    const _Float16* V2P = wp + 11264;
    const _Float16* V3P = wp + 15360;

    // ---- layer 1: feat[16,96] @ w1[96,64] + b1, relu -> LDS ----
    {
        v8f acc[4] = {};
        wave_mm<3, 4>(A, W1P, lane, acc);
        lds_fence();
#pragma unroll
        for (int nt = 0; nt < 4; ++nt) {
            const float bv = b1[nt * 16 + n];
#pragma unroll
            for (int r = 0; r < 8; ++r) {
                float v = acc[nt][r] + bv;
                v = v > 0.0f ? v : 0.0f;
                A[(mhi + r) * LDSTRIDE + nt * 16 + n] = (_Float16)v;
            }
        }
        lds_fence();
    }

    // ---- layer 2: h1[16,64] @ w2[64,16] + b2, relu; build z; emit sigma ----
    {
        v8f acc[1] = {};
        wave_mm<2, 1>(A, W2P, lane, acc);
        lds_fence();
        // positional encoding of d -> z cols [0,27); zero pad cols [42,64)
        if (lane < 16) {
            const float d0 = dvec[gp * 3 + 0], d1 = dvec[gp * 3 + 1], d2 = dvec[gp * 3 + 2];
            _Float16* zr = A + p * LDSTRIDE;
            zr[0] = (_Float16)d0; zr[1] = (_Float16)d1; zr[2] = (_Float16)d2;
#pragma unroll
            for (int j = 0; j < 4; ++j) {
                const float s = (float)(1 << j);
                zr[3 + 6 * j + 0] = (_Float16)__sinf(s * d0);
                zr[3 + 6 * j + 1] = (_Float16)__sinf(s * d1);
                zr[3 + 6 * j + 2] = (_Float16)__sinf(s * d2);
                zr[6 + 6 * j + 0] = (_Float16)__cosf(s * d0);
                zr[6 + 6 * j + 1] = (_Float16)__cosf(s * d1);
                zr[6 + 6 * j + 2] = (_Float16)__cosf(s * d2);
            }
#pragma unroll
            for (int k = 42; k < 64; ++k) zr[k] = (_Float16)0.0f;
        }
        // h[:, :15] -> z cols [27,42);  h[:,15] -> sigma (masked)
        const float bv = b2[n];
#pragma unroll
        for (int r = 0; r < 8; ++r) {
            float v = acc[0][r] + bv;
            v = v > 0.0f ? v : 0.0f;
            const int m = mhi + r;
            if (n < 15) {
                A[m * LDSTRIDE + 27 + n] = (_Float16)v;
            } else {
                const long g = gbase + m;
                if (g < nB) outS[g] = (Msk[m] != 0.0f) ? v : 0.0f;
            }
        }
        lds_fence();
    }

    // ---- layer 3: z[16,64] @ v1p[64,64] + c1, relu -> LDS ----
    {
        v8f acc[4] = {};
        wave_mm<2, 4>(A, V1P, lane, acc);
        lds_fence();
#pragma unroll
        for (int nt = 0; nt < 4; ++nt) {
            const float bv = c1[nt * 16 + n];
#pragma unroll
            for (int r = 0; r < 8; ++r) {
                float v = acc[nt][r] + bv;
                v = v > 0.0f ? v : 0.0f;
                A[(mhi + r) * LDSTRIDE + nt * 16 + n] = (_Float16)v;
            }
        }
        lds_fence();
    }

    // ---- layer 4: [16,64] @ v2p[64,64] + c2, relu -> LDS ----
    {
        v8f acc[4] = {};
        wave_mm<2, 4>(A, V2P, lane, acc);
        lds_fence();
#pragma unroll
        for (int nt = 0; nt < 4; ++nt) {
            const float bv = c2[nt * 16 + n];
#pragma unroll
            for (int r = 0; r < 8; ++r) {
                float v = acc[nt][r] + bv;
                v = v > 0.0f ? v : 0.0f;
                A[(mhi + r) * LDSTRIDE + nt * 16 + n] = (_Float16)v;
            }
        }
        lds_fence();
    }

    // ---- layer 5: [16,64] @ v3p[64,16(pad of 3)] + c3, sigmoid, mask -> outC ----
    {
        v8f acc[1] = {};
        wave_mm<2, 1>(A, V3P, lane, acc);
        if (n < 3) {
            const float bv = c3[n];
#pragma unroll
            for (int r = 0; r < 8; ++r) {
                const int  m = mhi + r;
                const long g = gbase + m;
                if (g < nB) {
                    const float v = acc[0][r] + bv;
                    const float s = 1.0f / (1.0f + __expf(-v));
                    outC[g * 3 + n] = (Msk[m] != 0.0f) ? s : 0.0f;
                }
            }
        }
    }
}

extern "C" void kernel_launch(void* const* d_in, const int* in_sizes, int n_in,
                              void* d_out, int out_size, void* d_ws, size_t ws_size,
                              hipStream_t stream) {
    const float* x  = (const float*)d_in[0];
    const float* d  = (const float*)d_in[1];
    const float* xy = (const float*)d_in[2];
    const float* yz = (const float*)d_in[3];
    const float* xz = (const float*)d_in[4];
    const float* w1 = (const float*)d_in[5];
    const float* b1 = (const float*)d_in[6];
    const float* w2 = (const float*)d_in[7];
    const float* b2 = (const float*)d_in[8];
    const float* v1 = (const float*)d_in[9];
    const float* c1 = (const float*)d_in[10];
    const float* v2 = (const float*)d_in[11];
    const float* c2 = (const float*)d_in[12];
    const float* v3 = (const float*)d_in[13];
    const float* c3 = (const float*)d_in[14];
    const int nB = in_sizes[0] / 3;

    _Float16* wp = (_Float16*)d_ws;   // 16384 f16 = 32 KB of packed weights
    float* outC = (float*)d_out;
    float* outS = outC + (long)nB * 3;

    // Pack weights into WMMA B-fragment layout (tiny; re-done each launch so
    // the call is stateless/deterministic).
    pack_weights<<<(6144 + 255) / 256, 256, 0, stream>>>(w1, 96, 64, 3, 4, wp + 0);
    pack_weights<<<(1024 + 255) / 256, 256, 0, stream>>>(w2, 64, 16, 2, 1, wp + 6144);
    pack_weights<<<(4096 + 255) / 256, 256, 0, stream>>>(v1, 42, 64, 2, 4, wp + 7168);
    pack_weights<<<(4096 + 255) / 256, 256, 0, stream>>>(v2, 64, 64, 2, 4, wp + 11264);
    pack_weights<<<(1024 + 255) / 256, 256, 0, stream>>>(v3, 64,  3, 2, 1, wp + 15360);

    const long nwaves = ((long)nB + 15) / 16;
    const int  blocks = (int)((nwaves + WAVES - 1) / WAVES);
    nerf_fused<<<blocks, 256, 0, stream>>>(x, d, xy, yz, xz, b1, b2, c1, c2, c3,
                                           wp, outC, outS, nB);
}
